// KnowformerLayer_79164837200214
// MI455X (gfx1250) — matrix operand order repaired and checked by reading, not compiled, not run on hardware
//
#include <hip/hip_runtime.h>
#include <math.h>
#include <stdint.h>

// ---------------------------------------------------------------------------
// Knowformer layer for MI455X (gfx1250, wave32, WMMA).
// Sizes fixed by reference: B=2, V=40000, D=128, R=32, H=4, L=2, E=400000.
// All dense GEMMs are (80000 x 128) @ (128 x 128): HBM-bound (~82MB/pass at
// 23.3 TB/s ~ 3.5us), so one-pass kernels with LDS-resident bf16 weights and
// v_wmma_f32_16x16x32_bf16 accumulation in f32.  A-tiles staged with
// GLOBAL_LOAD_ASYNC_TO_LDS_B128 when the toolchain exposes the builtin.
// ---------------------------------------------------------------------------

#define B_ 2
#define V_ 40000
#define D_ 128
#define R_ 32
#define H_ 4
#define L_ 2
#define E_ 400000
#define N_ (B_ * V_)            // 80000 rows
#define ND_ (N_ * D_)           // 10,240,000 elements per activation tensor

#if defined(__gfx1250__) && \
    __has_builtin(__builtin_amdgcn_global_load_async_to_lds_b128) && \
    __has_builtin(__builtin_amdgcn_s_wait_asynccnt)
#define HAVE_ASYNC_LDS 1
#endif

typedef __bf16 v16bf __attribute__((ext_vector_type(16)));
typedef float  v8f   __attribute__((ext_vector_type(8)));
typedef int    v4i_  __attribute__((ext_vector_type(4)));

#ifdef HAVE_ASYNC_LDS
typedef __attribute__((address_space(1))) v4i_* g_v4i_ptr;   // global (AS1)
typedef __attribute__((address_space(3))) v4i_* l_v4i_ptr;   // LDS    (AS3)
#endif

union BF16x16 { v16bf v; unsigned u[8]; };

__device__ __forceinline__ unsigned short f2bf(float f) {
    unsigned u = __float_as_uint(f);
    u += 0x7FFFu + ((u >> 16) & 1u);        // round-to-nearest-even
    return (unsigned short)(u >> 16);
}

// ---------------------------------------------------------------------------
// Generic GEMM: Out[N x 128] = act( A[N x 128] @ W[128 x 128] + bias
//                                   [+ indicator column-sum for head rows] )
// Block: 256 threads (8 waves), 64 rows x 128 cols per block.
// Wave w owns column tile [16w, 16w+16) and iterates 4 row tiles of 16.
// ---------------------------------------------------------------------------
#define LDW 130   // ushort LDS stride: even (4B-aligned pairs), conflict-free

__global__ __launch_bounds__(256) void gemm128_wmma(
    const float* __restrict__ A, const float* __restrict__ W,
    const float* __restrict__ bias, float* __restrict__ Out,
    int N, int relu, const float* __restrict__ specialAdd,
    const int* __restrict__ hIdx)
{
    __shared__ __attribute__((aligned(16))) unsigned short sW[128 * LDW];
    __shared__ __attribute__((aligned(16))) unsigned short sA[64 * LDW];
    __shared__ __attribute__((aligned(16))) float sAf[64 * 128];

    const int t = threadIdx.x;
    const int rowBase = blockIdx.x * 64;

    // Stage W transposed into LDS as bf16: sW[n*LDW + k] = W[k*128 + n]
    #pragma unroll 4
    for (int i = 0; i < 64; ++i) {
        int id = t + i * 256;                 // 16384 elements
        int k = id >> 7, n = id & 127;
        sW[n * LDW + k] = f2bf(W[id]);
    }

    // Stage the A tile (64 x 128 f32 = one contiguous 32KB block) into LDS.
#ifdef HAVE_ASYNC_LDS
    if (rowBase + 64 <= N) {
        const char* gsrc = (const char*)(A + (size_t)rowBase * 128);
        char* ldst = (char*)sAf;
        #pragma unroll
        for (int i = 0; i < 8; ++i) {
            unsigned off = (unsigned)(t + i * 256) * 16u;   // 16B per lane
            __builtin_amdgcn_global_load_async_to_lds_b128(
                (g_v4i_ptr)(uintptr_t)(gsrc + off),
                (l_v4i_ptr)(unsigned)(uintptr_t)(ldst + off),
                0, 0);
        }
        __builtin_amdgcn_s_wait_asynccnt(0);
    } else
#endif
    {
        #pragma unroll 4
        for (int i = 0; i < 32; ++i) {
            int id = t + i * 256;             // 8192 elements
            int r = id >> 7, c = id & 127;
            int gr = rowBase + r;
            sAf[id] = (gr < N) ? A[gr * 128 + c] : 0.0f;
        }
    }
    __syncthreads();

    // Pack A tile f32 -> bf16 (sA), LDS-to-LDS, cheap vs HBM.
    #pragma unroll 4
    for (int i = 0; i < 32; ++i) {
        int id = t + i * 256;
        int r = id >> 7, c = id & 127;
        sA[r * LDW + c] = f2bf(sAf[id]);
    }
    __syncthreads();

    const int wave = t >> 5, lane = t & 31;
    const int hf = lane >> 4;                 // lane half (0/1)
    const int nn = wave * 16 + (lane & 15);   // output column

    // B fragments (32x16 per K-step). Contiguous k order (same as A): lanes 0-15 hold K=0-15,
    // lanes 16-31 hold K=16-31; VGPR j packs K=2j,2j+1.
    const unsigned* Wrow = (const unsigned*)(sW + nn * LDW);
    v16bf bfrag[4];
    #pragma unroll
    for (int ks = 0; ks < 4; ++ks) {
        BF16x16 bu;
        int kb = (ks * 32 + hf * 16) >> 1;    // pair index
        #pragma unroll
        for (int i = 0; i < 8; ++i) bu.u[i] = Wrow[kb + i];
        bfrag[ks] = bu.v;
    }
    const float bv = bias[nn];
    const float sv = specialAdd ? specialAdd[nn] : 0.0f;

    for (int rt = 0; rt < 4; ++rt) {
        // A fragment (16x32): lane row m=lane&15; VGPR 0-7: K=hf*16+{0..15}
        // (matches B).  Hoist all 4 K-step fragments so the
        // 4 WMMAs issue back-to-back behind a single dscnt wait.
        const unsigned* Arow =
            (const unsigned*)(sA + (rt * 16 + (lane & 15)) * LDW);
        BF16x16 au[4];
        #pragma unroll
        for (int ks = 0; ks < 4; ++ks) {
            int ka = (ks * 32 + hf * 16) >> 1;
            #pragma unroll
            for (int i = 0; i < 4; ++i) au[ks].u[i]     = Arow[ka + i];
            #pragma unroll
            for (int i = 0; i < 4; ++i) au[ks].u[4 + i] = Arow[ka + 4 + i];
        }
        v8f acc = {0.f, 0.f, 0.f, 0.f, 0.f, 0.f, 0.f, 0.f};
        #pragma unroll
        for (int ks = 0; ks < 4; ++ks) {
            acc = __builtin_amdgcn_wmma_f32_16x16x32_bf16(
                false, au[ks].v, false, bfrag[ks], (short)0, acc,
                false, false);
        }
        // Epilogue: C/D layout — VGPR i: M = i + 8*hf, N = lane&15.
        #pragma unroll
        for (int i = 0; i < 8; ++i) {
            int rowG = rowBase + rt * 16 + i + 8 * hf;
            if (rowG >= N) continue;
            float val = acc[i] + bv;
            if (specialAdd) {                  // one-hot head-entity indicator
                int bb = rowG / V_;
                int vv = rowG - bb * V_;
                if (vv == hIdx[bb]) val += sv;
            }
            if (relu) val = fmaxf(val, 0.0f);
            Out[rowG * 128 + nn] = val;
        }
    }
}

// Column sums of the indicator half of fcvx_w1 (rows 128..255).
__global__ void colsum_kernel(const float* __restrict__ W1,
                              float* __restrict__ out)
{
    int n = threadIdx.x;                      // 128 threads
    float s = 0.f;
    for (int k = 0; k < 128; ++k) s += W1[(128 + k) * 128 + n];
    out[n] = s;
}

// zf[b, r, d] = bias + z[b] @ fcz_w  (B x 4096 outputs, tiny)
__global__ __launch_bounds__(256) void zf_kernel(
    const float* __restrict__ z, const float* __restrict__ W,
    const float* __restrict__ bias, float* __restrict__ out)
{
    int idx = blockIdx.x * 256 + threadIdx.x;   // [0, B*4096)
    int b = idx >> 12, j = idx & 4095;
    float acc = bias[j];
    for (int k = 0; k < 128; ++k) acc += z[b * 128 + k] * W[k * 4096 + j];
    out[idx] = acc;
}

// agg init: agg = beta[d] * vx   (fuses the "+ beta*vx" of the rspmm layer)
__global__ __launch_bounds__(256) void scale_init_kernel(
    const float* __restrict__ vx, const float* __restrict__ beta,
    float* __restrict__ out, int total)
{
    int i = blockIdx.x * 256 + threadIdx.x;
    if (i < total) out[i] = beta[i & 127] * vx[i];
}

// Relational sparse message passing: one block per 8 edges; thread t handles
// (b = t>>7, d = t&127).  agg[b,head,d] += zf[b,rel,d] * vx[b,tail,d]
#define EPB 8
__global__ __launch_bounds__(256) void rspmm_scatter(
    const int* __restrict__ eh, const int* __restrict__ er,
    const int* __restrict__ et, const float* __restrict__ zf,
    const float* __restrict__ vx, float* __restrict__ agg, int E)
{
    int t = threadIdx.x;
    int b = t >> 7, d = t & 127;
    int e0 = blockIdx.x * EPB;
    for (int i = 0; i < EPB; ++i) {
        int e = e0 + i;
        if (e >= E) break;
        int hh = eh[e], rr = er[e], tt = et[e];
        if (i + 1 < EPB && e + 1 < E)          // gfx1250 global_prefetch_b8
            __builtin_prefetch(&vx[(b * V_ + et[e + 1]) * 128 + d], 0, 1);
        float val = zf[b * 4096 + rr * 128 + d] * vx[(b * V_ + tt) * 128 + d];
        __hip_atomic_fetch_add(&agg[(b * V_ + hh) * 128 + d], val,
                               __ATOMIC_RELAXED, __HIP_MEMORY_SCOPE_AGENT);
    }
}

// LayerNorm over D=128: wave per row (wave32 shuffle reductions).
// dst = LN(s1 [+ s2]) * g + bta [+ res]
__global__ __launch_bounds__(256) void ln_kernel(
    const float* __restrict__ s1, const float* __restrict__ s2,
    const float* __restrict__ g, const float* __restrict__ bta,
    const float* __restrict__ res, float* __restrict__ dst, int N)
{
    int wave = threadIdx.x >> 5, lane = threadIdx.x & 31;
    int row = blockIdx.x * 8 + wave;
    if (row >= N) return;
    const int base = row * 128;
    float x[4], s = 0.f, sq = 0.f;
    #pragma unroll
    for (int i = 0; i < 4; ++i) {
        int d = lane + 32 * i;
        float v = s1[base + d] + (s2 ? s2[base + d] : 0.f);
        x[i] = v; s += v; sq += v * v;
    }
    for (int o = 16; o; o >>= 1) {
        s  += __shfl_xor(s,  o, 32);
        sq += __shfl_xor(sq, o, 32);
    }
    float mean = s * (1.0f / 128.0f);
    float var  = sq * (1.0f / 128.0f) - mean * mean;
    float inv  = rsqrtf(var + 1e-5f);
    #pragma unroll
    for (int i = 0; i < 4; ++i) {
        int d = lane + 32 * i;
        float y = (x[i] - mean) * inv * g[d] + bta[d]
                + (res ? res[base + d] : 0.f);
        dst[base + d] = y;
    }
}

__global__ __launch_bounds__(256) void zero_kernel(float* p, int n)
{
    int i = blockIdx.x * 256 + threadIdx.x;
    if (i < n) p[i] = 0.0f;
}

// Linear-attention reduction: kvs[b,h] = sum_v kn(v)^T v(v) (32x32),
// ksum = sum kn, vsum = sum v.  Chunked over V, atomic-accumulated.
__global__ __launch_bounds__(256) void attn_reduce(
    const float* __restrict__ K, const float* __restrict__ Vv,
    float* __restrict__ kvs, float* __restrict__ ksum,
    float* __restrict__ vsum, int chunks)
{
    __shared__ float skr[8][32];
    __shared__ float svr[8][32];
    __shared__ float snrm[8];
    const int bh = blockIdx.x / chunks;
    const int chunk = blockIdx.x % chunks;
    const int b = bh >> 2, h = bh & 3;
    const int rc = (V_ + chunks - 1) / chunks;
    const int start = chunk * rc;
    int end = start + rc; if (end > V_) end = V_;
    const int t = threadIdx.x;
    const int d0 = t >> 3;            // kvs row this thread owns
    const int j0 = (t & 7) * 4;       // 4 kvs cols this thread owns
    float accKV[4] = {0.f, 0.f, 0.f, 0.f};
    float accKS = 0.f;
    float accVS[4] = {0.f, 0.f, 0.f, 0.f};

    for (int base = start; base < end; base += 8) {
        #pragma unroll
        for (int j = 0; j < 2; ++j) {
            int id = t + j * 256;          // 8 rows x 64 (k|v) values
            int r8 = id >> 6, col = id & 63;
            int r = base + r8;
            float val = 0.f;
            if (r < end) {
                int off = (b * V_ + r) * 128 + h * 32;
                val = (col < 32) ? K[off + col] : Vv[off + col - 32];
            }
            if (col < 32) skr[r8][col] = val; else svr[r8][col - 32] = val;
        }
        __syncthreads();
        if (t < 8) {
            float s = 0.f;
            for (int j = 0; j < 32; ++j) s += skr[t][j] * skr[t][j];
            snrm[t] = 1.0f / fmaxf(sqrtf(s), 1e-12f);
        }
        __syncthreads();
        #pragma unroll
        for (int r8 = 0; r8 < 8; ++r8) {
            float kn = skr[r8][d0] * snrm[r8];
            #pragma unroll
            for (int jj = 0; jj < 4; ++jj) accKV[jj] += kn * svr[r8][j0 + jj];
            if ((t & 7) == 0) accKS += kn;
            if (t < 8) {
                #pragma unroll
                for (int jj = 0; jj < 4; ++jj) accVS[jj] += svr[r8][j0 + jj];
            }
        }
        __syncthreads();
    }
    #pragma unroll
    for (int jj = 0; jj < 4; ++jj)
        __hip_atomic_fetch_add(&kvs[bh * 1024 + d0 * 32 + j0 + jj], accKV[jj],
                               __ATOMIC_RELAXED, __HIP_MEMORY_SCOPE_AGENT);
    if ((t & 7) == 0)
        __hip_atomic_fetch_add(&ksum[bh * 32 + d0], accKS,
                               __ATOMIC_RELAXED, __HIP_MEMORY_SCOPE_AGENT);
    if (t < 8)
        for (int jj = 0; jj < 4; ++jj)
            __hip_atomic_fetch_add(&vsum[bh * 32 + j0 + jj], accVS[jj],
                                   __ATOMIC_RELAXED, __HIP_MEMORY_SCOPE_AGENT);
}

// Apply: out = (qn@kvs + vsum + v*V) / (qn.ksum + 2V); wave per (row,head).
// Writes back over Q in place.
__global__ __launch_bounds__(256) void attn_apply(
    float* __restrict__ Q, const float* __restrict__ Vv,
    const float* __restrict__ kvs, const float* __restrict__ ksum,
    const float* __restrict__ vsum)
{
    int wave = threadIdx.x >> 5, lane = threadIdx.x & 31;
    int s = blockIdx.x * 8 + wave;
    if (s >= N_ * H_) return;
    int row = s >> 2, h = s & 3;
    int b = row / V_;
    int bh = b * H_ + h;
    int off = row * 128 + h * 32 + lane;
    float qv = Q[off];
    float ss = qv * qv;
    for (int o = 16; o; o >>= 1) ss += __shfl_xor(ss, o, 32);
    float qn = qv / fmaxf(sqrtf(ss), 1e-12f);
    float p = qn * ksum[bh * 32 + lane];
    for (int o = 16; o; o >>= 1) p += __shfl_xor(p, o, 32);
    float acc = 0.f;
    const float* kv = kvs + bh * 1024;
    #pragma unroll 4
    for (int d = 0; d < 32; ++d) {
        float qd = __shfl(qn, d, 32);
        acc += qd * kv[d * 32 + lane];
    }
    float vv = Vv[off];
    float num = acc + vsum[bh * 32 + lane] + vv * (float)V_;
    float den = p + 2.0f * (float)V_;
    Q[off] = num / den;
}

// ---------------------------------------------------------------------------
extern "C" void kernel_launch(void* const* d_in, const int* in_sizes, int n_in,
                              void* d_out, int out_size, void* d_ws,
                              size_t ws_size, hipStream_t stream)
{
    (void)in_sizes; (void)n_in; (void)out_size; (void)ws_size;
    const float* x        = (const float*)d_in[0];
    const float* z        = (const float*)d_in[1];
    const int*   h_index  = (const int*)d_in[2];
    const int*   edge_head= (const int*)d_in[3];
    const int*   edge_rel = (const int*)d_in[4];
    const int*   edge_tail= (const int*)d_in[5];
    const float* fcvx_w1  = (const float*)d_in[6];
    const float* fcvx_b1  = (const float*)d_in[7];
    const float* fcvx_w2  = (const float*)d_in[8];
    const float* fcvx_b2  = (const float*)d_in[9];
    const float* vl_fcz_w = (const float*)d_in[10];
    const float* vl_fcz_b = (const float*)d_in[11];
    const float* vl_beta  = (const float*)d_in[12];
    const float* vl_fco_w1= (const float*)d_in[13];
    const float* vl_fco_b1= (const float*)d_in[14];
    const float* vl_fco_w2= (const float*)d_in[15];
    const float* vl_fco_b2= (const float*)d_in[16];
    const float* vl_ln_g  = (const float*)d_in[17];
    const float* vl_ln_b  = (const float*)d_in[18];
    const float* wq_w = (const float*)d_in[19];
    const float* wq_b = (const float*)d_in[20];
    const float* wk_w = (const float*)d_in[21];
    const float* wk_b = (const float*)d_in[22];
    const float* wv_w = (const float*)d_in[23];
    const float* wv_b = (const float*)d_in[24];
    const float* an_g = (const float*)d_in[25];
    const float* an_b = (const float*)d_in[26];
    const float* ffn_w1 = (const float*)d_in[27];
    const float* ffn_b1 = (const float*)d_in[28];
    const float* ffn_w2 = (const float*)d_in[29];
    const float* ffn_b2 = (const float*)d_in[30];
    const float* n_g  = (const float*)d_in[31];
    const float* n_b  = (const float*)d_in[32];
    float* out = (float*)d_out;

    // Workspace layout (floats): 4 activation buffers + small scratch (~164MB)
    float* ws     = (float*)d_ws;
    float* vx     = ws;
    float* bufA   = ws + (size_t)ND_;
    float* bufB   = ws + 2 * (size_t)ND_;
    float* bufC   = ws + 3 * (size_t)ND_;
    float* zf     = ws + 4 * (size_t)ND_;          // B*R*D = 8192
    float* kvs    = zf + B_ * R_ * D_;             // B*H*32*32 = 8192
    float* ksum   = kvs + 8192;                    // 256
    float* vsum   = ksum + 256;                    // 256
    float* colsum = vsum + 256;                    // 128

    const int gemmBlocks = (N_ + 63) / 64;         // 1250
    const int lnBlocks   = (N_ + 7) / 8;           // 10000
    const int ewBlocks   = (ND_ + 255) / 256;      // 40000

    // --- fc_v_x MLP with head-entity indicator ---
    colsum_kernel<<<1, 128, 0, stream>>>(fcvx_w1, colsum);
    gemm128_wmma<<<gemmBlocks, 256, 0, stream>>>(
        x, fcvx_w1, fcvx_b1, bufA, N_, 1, colsum, h_index);
    gemm128_wmma<<<gemmBlocks, 256, 0, stream>>>(
        bufA, fcvx_w2, fcvx_b2, vx, N_, 0, nullptr, nullptr);

    // --- L stacked relational message-passing layers ---
    for (int l = 0; l < L_; ++l) {
        zf_kernel<<<(B_ * 4096) / 256, 256, 0, stream>>>(
            z, vl_fcz_w + (size_t)l * D_ * D_ * R_, vl_fcz_b + l * D_ * R_, zf);
        scale_init_kernel<<<ewBlocks, 256, 0, stream>>>(
            vx, vl_beta + l * D_, bufA, ND_);
        rspmm_scatter<<<(E_ + EPB - 1) / EPB, 256, 0, stream>>>(
            edge_head, edge_rel, edge_tail, zf, vx, bufA, E_);
        gemm128_wmma<<<gemmBlocks, 256, 0, stream>>>(
            bufA, vl_fco_w1 + l * D_ * D_, vl_fco_b1 + l * D_, bufB,
            N_, 1, nullptr, nullptr);
        gemm128_wmma<<<gemmBlocks, 256, 0, stream>>>(
            bufB, vl_fco_w2 + l * D_ * D_, vl_fco_b2 + l * D_, bufA,
            N_, 0, nullptr, nullptr);
        // vx = LN(t)*g + b + vx
        ln_kernel<<<lnBlocks, 256, 0, stream>>>(
            bufA, nullptr, vl_ln_g + l * D_, vl_ln_b + l * D_, vx, vx, N_);
    }

    // --- linear attention ---
    gemm128_wmma<<<gemmBlocks, 256, 0, stream>>>(
        vx, wq_w, wq_b, bufA, N_, 0, nullptr, nullptr);   // q
    gemm128_wmma<<<gemmBlocks, 256, 0, stream>>>(
        vx, wk_w, wk_b, bufB, N_, 0, nullptr, nullptr);   // k
    gemm128_wmma<<<gemmBlocks, 256, 0, stream>>>(
        vx, wv_w, wv_b, bufC, N_, 0, nullptr, nullptr);   // v
    zero_kernel<<<(8192 + 256 + 256 + 255) / 256, 256, 0, stream>>>(
        kvs, 8192 + 256 + 256);
    const int chunks = 80;                                // 500 rows/chunk
    attn_reduce<<<B_ * H_ * chunks, 256, 0, stream>>>(
        bufB, bufC, kvs, ksum, vsum, chunks);
    attn_apply<<<(N_ * H_) / 8, 256, 0, stream>>>(
        bufA, bufC, kvs, ksum, vsum);                     // attn out -> bufA
    // h = LN(vx + attn) * an_g + an_b -> bufB
    ln_kernel<<<lnBlocks, 256, 0, stream>>>(
        vx, bufA, an_g, an_b, nullptr, bufB, N_);

    // --- FFN + final LN ---
    gemm128_wmma<<<gemmBlocks, 256, 0, stream>>>(
        bufB, ffn_w1, ffn_b1, bufC, N_, 1, nullptr, nullptr);
    gemm128_wmma<<<gemmBlocks, 256, 0, stream>>>(
        bufC, ffn_w2, ffn_b2, bufA, N_, 0, nullptr, nullptr);
    ln_kernel<<<lnBlocks, 256, 0, stream>>>(
        bufB, bufA, n_g, n_b, nullptr, out, N_);
}
